// NetFull_89919435309225
// MI455X (gfx1250) — matrix operand order repaired and checked by reading, not compiled, not run on hardware
//
#include <hip/hip_runtime.h>
#include <math.h>
#include <stdint.h>

// ---------------------------------------------------------------------------
// CDNA5 / gfx1250 implementation of the KPConv-style point network.
// All dense math goes through v_wmma_f32_16x16x32_f16 (f16 in, f32 acc).
// The KPConv einsum('ef,ek,kfh->eh') is computed as (f (x) w) @ W1 with the
// A-operand generated on the fly inside the GEMM tile loader (AMODE=1).
// B tiles are staged TRANSPOSED in LDS so both A and B fragments are
// contiguous 16B-aligned ds_load_b128 reads. Stem weights are broadcast to
// LDS via the Tensor Data Mover (tensor_load_to_lds + s_wait_tensorcnt).
// ---------------------------------------------------------------------------

#define BN_INV 0.99999500003749975f   // 1/sqrt(1+1e-5)

typedef __attribute__((ext_vector_type(16))) _Float16 v16h;
typedef __attribute__((ext_vector_type(8)))  float    v8f;
typedef __attribute__((ext_vector_type(4)))  unsigned vu4;
typedef __attribute__((ext_vector_type(4)))  int      vi4;
typedef __attribute__((ext_vector_type(8)))  int      vi8;

__device__ __forceinline__ float lrelu_f(float x) { return x >= 0.f ? x : 0.01f * x; }

enum { EP_BIAS = 0, EP_LRELU = 1, EP_LRELU_BN = 2, EP_BN = 3, EP_BN_LRELU = 4, EP_BN_ADD_LRELU = 5 };

// ---------------------------------------------------------------------------
// Generic tiled WMMA GEMM:  C[M,N] = epilogue(A[M,K] @ B[K,N] + bias)
// AMODE==0: A read from memory (fp32 row-major).
// AMODE==1: A[e, kk] = Aw[e, kk/F] * A[e, kk%F]  (KPConv expanded operand).
// Tile: 128x64 per workgroup (256 thr = 8 waves, 4x2 wave grid, 32x32/wave).
// ---------------------------------------------------------------------------
template <int AMODE>
__global__ __launch_bounds__(256) void wmma_gemm(
    const float* __restrict__ A, const float* __restrict__ Aw, int F,
    const float* __restrict__ B, const float* __restrict__ bias,
    const float* __restrict__ g, const float* __restrict__ beta,
    const float* __restrict__ res, float* __restrict__ C,
    int M, int N, int K, int mode)
{
    __shared__ _Float16 As [128][40];  // 128 x 32 (+8 pad), row stride 80B (16B mult)
    __shared__ _Float16 BsT[64][56];   // 64 cols x 32 k (transposed), stride 112B (16B mult)

    const int tid  = threadIdx.x;
    const int lane = tid & 31;
    const int wave = tid >> 5;
    const int wm   = wave & 3;         // 0..3  -> rows
    const int wn   = wave >> 2;        // 0..1  -> cols
    const int bM   = blockIdx.y * 128;
    const int bN   = blockIdx.x * 64;
    const int hl   = lane >> 4;        // half-wave select
    const int l16  = lane & 15;

    v8f acc[2][2];
    #pragma unroll
    for (int i = 0; i < 2; ++i)
        #pragma unroll
        for (int j = 0; j < 2; ++j)
            acc[i][j] = v8f{0.f, 0.f, 0.f, 0.f, 0.f, 0.f, 0.f, 0.f};

    for (int k0 = 0; k0 < K; k0 += 32) {
        // --- prefetch next K tile (global_prefetch_b8), branch-free clamp ---
        if (k0 + 32 < K) {
            int pr = bM + (tid >> 1); pr = pr < M ? pr : M - 1;
            if (AMODE == 0) __builtin_prefetch(&A[(size_t)pr * K + k0 + 32], 0, 1);
            else            __builtin_prefetch(&A[(size_t)pr * F + ((k0 + 32) % F)], 0, 1);
            int pk = k0 + 32 + (tid & 31); pk = pk < K ? pk : K - 1;
            __builtin_prefetch(&B[(size_t)pk * N + bN], 0, 1);
        }
        // --- stage A tile (128x32): 16 elems/thread, fp32 -> f16 ---
        #pragma unroll
        for (int i = 0; i < 16; ++i) {
            int lin = i * 256 + tid;
            int r = lin >> 5, c = lin & 31;
            int gr = bM + r, gk = k0 + c;
            float v = 0.f;
            if (gr < M && gk < K) {
                if (AMODE == 0) {
                    v = A[(size_t)gr * K + gk];
                } else {
                    int kp = gk / F, ff = gk - kp * F;
                    v = Aw[(size_t)gr * 16 + kp] * A[(size_t)gr * F + ff];
                }
            }
            As[r][c] = (_Float16)v;
        }
        // --- stage B tile (32x64) transposed into BsT[n][k] ---
        #pragma unroll
        for (int i = 0; i < 8; ++i) {
            int lin = i * 256 + tid;
            int r = lin >> 6, c = lin & 63;
            int gk = k0 + r, gc = bN + c;
            float v = (gk < K && gc < N) ? B[(size_t)gk * N + gc] : 0.f;
            BsT[c][r] = (_Float16)v;
        }
        __syncthreads();

        // --- build fragments per ISA 7.12.2 layouts and issue WMMA ---
        v16h afr[2], bfr[2];
        #pragma unroll
        for (int mr = 0; mr < 2; ++mr) {
            int row = wm * 32 + mr * 16 + l16;      // A: lane holds row m
            #pragma unroll
            for (int v = 0; v < 8; ++v) {
                int kk = ((v & 4) << 2) + ((v & 3) << 1) + (hl << 3);
                afr[mr][2 * v]     = As[row][kk];
                afr[mr][2 * v + 1] = As[row][kk + 1];
            }
        }
        #pragma unroll
        for (int nr = 0; nr < 2; ++nr) {
            int col = wn * 32 + nr * 16 + l16;      // B: lane holds column n
            #pragma unroll
            for (int v = 0; v < 16; ++v)            // two 16B runs (k=8h.., 16+8h..) -> 2x ds_load_b128
                bfr[nr][v] = BsT[col][(v & 8) * 2 + hl * 8 + (v & 7)];
        }
        #pragma unroll
        for (int mr = 0; mr < 2; ++mr)
            #pragma unroll
            for (int nr = 0; nr < 2; ++nr)
                acc[mr][nr] = __builtin_amdgcn_wmma_f32_16x16x32_f16(
                    false, afr[mr], false, bfr[nr], (short)0, acc[mr][nr], false, false);
        __syncthreads();
    }

    // --- fused epilogue + store (C/D layout: VGPR r -> M=r(+8 for hi lanes)) ---
    #pragma unroll
    for (int mr = 0; mr < 2; ++mr) {
        #pragma unroll
        for (int nr = 0; nr < 2; ++nr) {
            #pragma unroll
            for (int r = 0; r < 8; ++r) {
                int row = bM + wm * 32 + mr * 16 + r + hl * 8;
                int col = bN + wn * 32 + nr * 16 + l16;
                if (row < M && col < N) {
                    float v = acc[mr][nr][r];
                    if (bias) v += bias[col];
                    if (mode == EP_LRELU) {
                        v = lrelu_f(v);
                    } else if (mode == EP_LRELU_BN) {
                        v = lrelu_f(v); v = v * (g[col] * BN_INV) + beta[col];
                    } else if (mode == EP_BN) {
                        v = v * (g[col] * BN_INV) + beta[col];
                    } else if (mode == EP_BN_LRELU) {
                        v = lrelu_f(v * (g[col] * BN_INV) + beta[col]);
                    } else if (mode == EP_BN_ADD_LRELU) {
                        v = v * (g[col] * BN_INV) + beta[col];
                        v = lrelu_f(v + res[(size_t)row * N + col]);
                    }
                    C[(size_t)row * N + col] = v;
                }
            }
        }
    }
}

// ---------------------------------------------------------------------------
// Small scalar kernels (gathers, edge features, reductions, epilogues)
// ---------------------------------------------------------------------------
__global__ void build_pos4_k(const float* __restrict__ pos, const float* __restrict__ refl,
                             const float* __restrict__ sf, float* __restrict__ p4, int n) {
    int i = blockIdx.x * blockDim.x + threadIdx.x;
    if (i >= n) return;
    float inv = 1.0f / sf[0];
    p4[(size_t)i * 4 + 0] = pos[(size_t)i * 3 + 0] * inv;
    p4[(size_t)i * 4 + 1] = pos[(size_t)i * 3 + 1] * inv;
    p4[(size_t)i * 4 + 2] = pos[(size_t)i * 3 + 2] * inv;
    p4[(size_t)i * 4 + 3] = refl[i];
}

__global__ void gather_rows_k(const float* __restrict__ in, const int* __restrict__ idx,
                              float* __restrict__ out, int n, int C) {
    int t = blockIdx.x * blockDim.x + threadIdx.x;
    if (t >= n * C) return;
    int i = t / C, c = t - i * C;
    out[t] = in[(size_t)idx[i] * C + c];
}

// Stem: per-edge MLP 4->45 (lrelu) ->64 (lrelu, bn); output [E,64].
// The 45x64 weight matrix is broadcast into LDS via the Tensor Data Mover.
__global__ __launch_bounds__(256) void stem_edge_k(
    const float* __restrict__ p4,
    const int* __restrict__ src, const int* __restrict__ tgt,
    const float* __restrict__ W0, const float* __restrict__ b0,
    const float* __restrict__ W1, const float* __restrict__ b1,
    const float* __restrict__ beta, const float* __restrict__ g,
    float* __restrict__ out, int E) {
    __shared__ float sW1[45 * 64];     // first shared object -> LDS offset 0
    __shared__ float sW0[4 * 45];
    __shared__ float sb0[45];
    __shared__ float sb1[64], sbeta[64], sg[64];

#if defined(__gfx1250__) && __has_builtin(__builtin_amdgcn_tensor_load_to_lds)
    if (threadIdx.x < 32) {            // one wave issues the TDM op
        unsigned long long ga = (unsigned long long)(uintptr_t)W1;
        // D# group0: count=1 | lds_addr=0 | global_addr | type=2 ("image")
        vu4 g0 = { 1u,
                   0u,
                   (unsigned)(ga & 0xffffffffu),
                   (unsigned)((ga >> 32) & 0x01ffffffu) | (2u << 30) };
        // D# group1: data_size=4B, tensor 2880x1, tile 2880x1, stride 2880
        vi8 g1 = { (int)(2u << 16),
                   (int)(2880u << 16),
                   (int)(1u << 16),
                   (int)(2880u << 16),
                   1,
                   2880,
                   0, 0 };
        vi4 gz = { 0, 0, 0, 0 };
  #if __clang_major__ >= 23
        vi8 gz8 = { 0, 0, 0, 0, 0, 0, 0, 0 };
        __builtin_amdgcn_tensor_load_to_lds(g0, g1, gz, gz, gz8, 0);
  #else
        __builtin_amdgcn_tensor_load_to_lds(g0, g1, gz, gz, 0);
  #endif
        __builtin_amdgcn_s_wait_tensorcnt(0);
    }
#else
    for (int i = threadIdx.x; i < 45 * 64; i += 256) sW1[i] = W1[i];
#endif
    for (int i = threadIdx.x; i < 180; i += 256) sW0[i] = W0[i];
    for (int i = threadIdx.x; i < 45; i += 256) sb0[i] = b0[i];
    for (int i = threadIdx.x; i < 64; i += 256) { sb1[i] = b1[i]; sbeta[i] = beta[i]; sg[i] = g[i]; }
    __syncthreads();

    int e = blockIdx.x * blockDim.x + threadIdx.x;
    if (e >= E) return;
    int s = src[e], t = tgt[e];
    float r[4];
    #pragma unroll
    for (int j = 0; j < 4; ++j) r[j] = p4[(size_t)s * 4 + j] - p4[(size_t)t * 4 + j];
    float h1[45];
    for (int j = 0; j < 45; ++j) {
        float a = sb0[j];
        #pragma unroll
        for (int i = 0; i < 4; ++i) a += r[i] * sW0[i * 45 + j];
        h1[j] = lrelu_f(a);
    }
    for (int j = 0; j < 64; ++j) {
        float a = sb1[j];
        for (int i = 0; i < 45; ++i) a += h1[i] * sW1[i * 64 + j];
        a = lrelu_f(a);
        a = a * (sg[j] * BN_INV) + sbeta[j];
        out[(size_t)e * 64 + j] = a;
    }
}

// segment_max over K contiguous edge rows per target (tgt = repeat(arange,K))
__global__ void segmax_k(const float* __restrict__ in, float* __restrict__ out,
                         int n_tgt, int C, int Kdeg) {
    int t = blockIdx.x * blockDim.x + threadIdx.x;
    if (t >= n_tgt * C) return;
    int i = t / C, c = t - i * C;
    float m = -3.402823466e38f;
    for (int k = 0; k < Kdeg; ++k)
        m = fmaxf(m, in[((size_t)i * Kdeg + k) * C + c]);
    if (!isfinite(m)) m = 0.f;
    out[t] = m;
}

// f[e, 0:C] = x[src[e]]; f[e, C:C+4] = rel; f[e, C+4] = |rel3|
__global__ void sa_feat_k(const float* __restrict__ x, const float* __restrict__ p4,
                          const float* __restrict__ p4ds,
                          const int* __restrict__ src, const int* __restrict__ tgt,
                          float* __restrict__ f, int E, int C) {
    int F = C + 5;
    int t = blockIdx.x * blockDim.x + threadIdx.x;
    if (t >= E * F) return;
    int e = t / F, c = t - e * F;
    int s = src[e], tg = tgt[e];
    if (c < C) { f[t] = x[(size_t)s * C + c]; return; }
    float rel[4];
    #pragma unroll
    for (int j = 0; j < 4; ++j) rel[j] = p4[(size_t)s * 4 + j] - p4ds[(size_t)tg * 4 + j];
    if (c < C + 4) f[t] = rel[c - C];
    else f[t] = sqrtf(rel[0]*rel[0] + rel[1]*rel[1] + rel[2]*rel[2] + 1e-12f);
}

// kernel-point influence weights w[e,k] = relu(1 - |rel3 - kp_k|) (SIGMA=1)
__global__ void sa_w_k(const float* __restrict__ p4, const float* __restrict__ p4ds,
                       const int* __restrict__ src, const int* __restrict__ tgt,
                       const float* __restrict__ kp, float* __restrict__ w, int E) {
    int t = blockIdx.x * blockDim.x + threadIdx.x;
    if (t >= E * 16) return;
    int e = t >> 4, k = t & 15;
    int s = src[e], tg = tgt[e];
    float dk = 1e-12f;
    #pragma unroll
    for (int j = 0; j < 3; ++j) {
        float r = p4[(size_t)s * 4 + j] - p4ds[(size_t)tg * 4 + j] - kp[k * 3 + j];
        dk += r * r;
    }
    w[t] = fmaxf(0.f, 1.f - sqrtf(dk));
}

// depthwise (k=1) + BN + lrelu:  h = lrelu(bn(h*dw + dwb, g, b))
__global__ void dwbn_k(float* __restrict__ h, const float* __restrict__ dw,
                       const float* __restrict__ dwb, const float* __restrict__ g,
                       const float* __restrict__ b, int total, int C) {
    int t = blockIdx.x * blockDim.x + threadIdx.x;
    if (t >= total) return;
    int c = t % C;
    float v = h[t] * dw[c] + dwb[c];
    v = v * (g[c] * BN_INV) + b[c];
    h[t] = lrelu_f(v);
}

__global__ void bn_k(float* __restrict__ h, const float* __restrict__ g,
                     const float* __restrict__ b, int total, int C) {
    int t = blockIdx.x * blockDim.x + threadIdx.x;
    if (t >= total) return;
    int c = t % C;
    h[t] = h[t] * (g[c] * BN_INV) + b[c];
}

__global__ void se_mean_k(const float* __restrict__ x, float* __restrict__ z, int n, int C) {
    int c = blockIdx.x * blockDim.x + threadIdx.x;
    if (c >= C) return;
    float s = 0.f;
    for (int i = 0; i < n; ++i) s += x[(size_t)i * C + c];
    z[c] = s / fmaxf((float)n, 1.f);
}

__global__ void se_mlp_k(const float* __restrict__ z, const float* __restrict__ W1,
                         const float* __restrict__ W2, float* __restrict__ s, int C) {
    __shared__ float hbuf[256];
    int h4 = C >> 2;
    for (int i = threadIdx.x; i < h4; i += blockDim.x) {
        float a = 0.f;
        for (int j = 0; j < C; ++j) a += z[j] * W1[(size_t)j * h4 + i];
        hbuf[i] = fmaxf(a, 0.f);
    }
    __syncthreads();
    for (int i = threadIdx.x; i < C; i += blockDim.x) {
        float a = 0.f;
        for (int j = 0; j < h4; ++j) a += hbuf[j] * W2[(size_t)j * C + i];
        s[i] = 1.f / (1.f + expf(-a));
    }
}

__global__ void scale_k(float* __restrict__ x, const float* __restrict__ s, int total, int C) {
    int t = blockIdx.x * blockDim.x + threadIdx.x;
    if (t >= total) return;
    x[t] *= s[t % C];
}

// out[n, 0:Ca] = a[gidx[n],:] ; out[n, Ca:Ca+Cb] = b[n,:]
__global__ void concat_k(const float* __restrict__ a, const int* __restrict__ gidx, int Ca,
                         const float* __restrict__ b, int Cb, float* __restrict__ out, int n) {
    int C = Ca + Cb;
    int t = blockIdx.x * blockDim.x + threadIdx.x;
    if (t >= n * C) return;
    int i = t / C, c = t - i * C;
    out[t] = (c < Ca) ? a[(size_t)gidx[i] * Ca + c] : b[(size_t)i * Cb + (c - Ca)];
}

// head logits: out[n,2] = h @ W2 + b2  (N=2 too narrow for WMMA)
__global__ void logits_k(const float* __restrict__ h, const float* __restrict__ W,
                         const float* __restrict__ b, float* __restrict__ out, int n) {
    int t = blockIdx.x * blockDim.x + threadIdx.x;
    if (t >= n * 2) return;
    int i = t >> 1, c = t & 1;
    float a = b[c];
    for (int j = 0; j < 1024; ++j) a += h[(size_t)i * 1024 + j] * W[j * 2 + c];
    out[t] = a;
}

// ---------------------------------------------------------------------------
// Host orchestration
// ---------------------------------------------------------------------------
struct MlpP  { const float *W0, *b0, *W1, *b1, *beta, *g; };
struct HeadP { const float *W1, *W2, *b1, *b2, *bb, *g; };
struct BlockP { const float *We, *Wp, *Wpw, *b1, *b2, *b3, *b4, *b5,
                *be, *bp, *bpw, *dw, *dwb, *g1, *g2, *g3, *g4, *g5; };
struct SaP { const float *W1, *W2, *b1, *b2, *be2, *g2, *kp, *seW1, *seW2;
             BlockP blk[6]; int nblk; };

static inline int cdiv(int a, int b) { return (a + b - 1) / b; }

static void launch_gemm(int amode, const float* A, const float* Aw, int F,
                        const float* B, const float* bias, const float* g,
                        const float* beta, const float* res, float* C,
                        int M, int N, int K, int mode, hipStream_t s) {
    dim3 grid(cdiv(N, 64), cdiv(M, 128));
    if (amode == 0)
        wmma_gemm<0><<<grid, 256, 0, s>>>(A, Aw, F, B, bias, g, beta, res, C, M, N, K, mode);
    else
        wmma_gemm<1><<<grid, 256, 0, s>>>(A, Aw, F, B, bias, g, beta, res, C, M, N, K, mode);
}

extern "C" void kernel_launch(void* const* d_in, const int* in_sizes, int n_in,
                              void* d_out, int out_size, void* d_ws, size_t ws_size,
                              hipStream_t stream) {
    // ---- data inputs (setup_inputs order) ----
    const float* pos  = (const float*)d_in[0];
    const float* refl = (const float*)d_in[1];
    const float* sf   = (const float*)d_in[2];
    const int* stem_src = (const int*)d_in[4];
    const int* stem_tgt = (const int*)d_in[5];
    const int* idx1 = (const int*)d_in[6];
    const int* e1s = (const int*)d_in[7],  * e1t = (const int*)d_in[8];
    const int* idx2 = (const int*)d_in[9];
    const int* e2s = (const int*)d_in[10], * e2t = (const int*)d_in[11];
    const int* idx3 = (const int*)d_in[12];
    const int* e3s = (const int*)d_in[13], * e3t = (const int*)d_in[14];
    const int* idx4 = (const int*)d_in[15];
    const int* e4s = (const int*)d_in[16], * e4t = (const int*)d_in[17];
    const int* fp4_idx = (const int*)d_in[18];
    const int* fp3_idx = (const int*)d_in[19];
    const int* fp2_idx = (const int*)d_in[20];
    const int* fp1_idx = (const int*)d_in[21];

    const int N0 = in_sizes[1];
    const int N1 = in_sizes[6], N2 = in_sizes[9], N3 = in_sizes[12], N4 = in_sizes[15];
    const int E0 = in_sizes[4];
    const int E1 = in_sizes[7], E2 = in_sizes[10], E3 = in_sizes[13], E4 = in_sizes[16];

    // ---- params: jax pytree leaf order (sorted dict keys, lists in order) ----
    int cur = 22;
    auto nf = [&]() { return (const float*)d_in[cur++]; };
    MlpP fp[4];   // fp1, fp2, fp3, fp4
    for (int i = 0; i < 4; ++i) {
        fp[i].W0 = nf(); fp[i].b0 = nf();
        fp[i].W1 = nf(); fp[i].b1 = nf(); fp[i].beta = nf(); fp[i].g = nf();
    }
    HeadP hd; hd.W1 = nf(); hd.W2 = nf(); hd.b1 = nf(); hd.b2 = nf(); hd.bb = nf(); hd.g = nf();
    const int nblk_arr[4] = {3, 6, 2, 1};
    SaP sa[4];
    for (int i = 0; i < 4; ++i) {
        sa[i].W1 = nf(); sa[i].W2 = nf(); sa[i].b1 = nf(); sa[i].b2 = nf(); sa[i].be2 = nf();
        sa[i].nblk = nblk_arr[i];
        for (int b = 0; b < sa[i].nblk; ++b) {
            BlockP& B = sa[i].blk[b];
            B.We = nf(); B.Wp = nf(); B.Wpw = nf();
            B.b1 = nf(); B.b2 = nf(); B.b3 = nf(); B.b4 = nf(); B.b5 = nf();
            B.be = nf(); B.bp = nf(); B.bpw = nf();
            B.dw = nf(); B.dwb = nf();
            B.g1 = nf(); B.g2 = nf(); B.g3 = nf(); B.g4 = nf(); B.g5 = nf();
        }
        sa[i].g2 = nf(); sa[i].kp = nf(); sa[i].seW1 = nf(); sa[i].seW2 = nf();
    }
    MlpP stem; stem.W0 = nf(); stem.b0 = nf();
    stem.W1 = nf(); stem.b1 = nf(); stem.beta = nf(); stem.g = nf();

    // ---- workspace arena ----
    size_t off = 0;
    auto alloc = [&](size_t nfloats) -> float* {
        float* p = (float*)((char*)d_ws + off);
        off += ((nfloats * sizeof(float)) + 255) & ~(size_t)255;
        return p;
    };
    float* p4_0 = alloc((size_t)N0 * 4);
    float* x0   = alloc((size_t)N0 * 64);
    float* pos1 = alloc((size_t)N1 * 3); float* refl1 = alloc(N1); float* p4_1 = alloc((size_t)N1 * 4);
    float* pos2 = alloc((size_t)N2 * 3); float* refl2 = alloc(N2); float* p4_2 = alloc((size_t)N2 * 4);
    float* pos3 = alloc((size_t)N3 * 3); float* refl3 = alloc(N3); float* p4_3 = alloc((size_t)N3 * 4);
    float* p4ds = alloc((size_t)N1 * 4);                 // reused per level (max n_tgt)
    const int couts[4] = {128, 256, 512, 1024};
    const int cins[4]  = {64, 128, 256, 512};
    const int hs[4]    = {90, 181, 362, 724};
    const int ntg[4]   = {N1, N2, N3, N4};
    float *xA[4], *xB[4];
    for (int i = 0; i < 4; ++i) { xA[i] = alloc((size_t)ntg[i] * couts[i]);
                                  xB[i] = alloc((size_t)ntg[i] * couts[i]); }
    float* zbuf = alloc(1024);
    float* sbuf = alloc(1024);
    const size_t SCR_F = 44ull * 1024 * 1024;   // 176MB each (fp1 concat = 43.52M floats)
    float* SCRA = alloc(SCR_F);
    float* SCRB = alloc(SCR_F);
    (void)ws_size; (void)n_in; (void)out_size;

    const int TB = 256;
    auto g1d = [&](int total) { return dim3(cdiv(total, TB)); };

    // ---- stem ----
    build_pos4_k<<<g1d(N0), TB, 0, stream>>>(pos, refl, sf, p4_0, N0);
    stem_edge_k<<<g1d(E0), TB, 0, stream>>>(p4_0, stem_src, stem_tgt,
        stem.W0, stem.b0, stem.W1, stem.b1, stem.beta, stem.g, SCRA, E0);
    segmax_k<<<g1d(N0 * 64), TB, 0, stream>>>(SCRA, x0, N0, 64, 8);

    // ---- SA stage runner ----
    auto run_sa = [&](const SaP& P, const float* xsrc, const float* p4src,
                      const int* idx, const int* esrc, const int* etgt,
                      int n_tgt, int E, int Cin, int H, int Cout,
                      float* xAb, float* xBb) -> float* {
        gather_rows_k<<<g1d(n_tgt * 4), TB, 0, stream>>>(p4src, idx, p4ds, n_tgt, 4);
        const int F = Cin + 5;
        float* fb = SCRA;
        size_t fsz = (((size_t)E * F * sizeof(float)) + 255) & ~(size_t)255;
        float* wb = (float*)((char*)SCRA + fsz);
        sa_feat_k<<<g1d(E * F), TB, 0, stream>>>(xsrc, p4src, p4ds, esrc, etgt, fb, E, Cin);
        sa_w_k<<<g1d(E * 16), TB, 0, stream>>>(p4src, p4ds, esrc, etgt, P.kp, wb, E);
        float* h1 = SCRB;
        launch_gemm(1, fb, wb, F, P.W1, P.b1, nullptr, nullptr, nullptr,
                    h1, E, H, 16 * F, EP_LRELU, stream);
        float* h2 = SCRA;   // f/w dead
        launch_gemm(0, h1, nullptr, 0, P.W2, P.b2, P.g2, P.be2, nullptr,
                    h2, E, Cout, H, EP_LRELU_BN, stream);
        segmax_k<<<g1d(n_tgt * Cout), TB, 0, stream>>>(h2, xAb, n_tgt, Cout, 16);
        float* xcur = xAb, *xalt = xBb;
        const int e4 = 4 * Cout;
        for (int bi = 0; bi < P.nblk; ++bi) {
            const BlockP& B = P.blk[bi];
            float* t1 = SCRB;
            float* t2 = SCRA;
            launch_gemm(0, xcur, nullptr, 0, B.We, B.be, B.g1, B.b1, nullptr,
                        t1, n_tgt, e4, Cout, EP_BN_LRELU, stream);
            dwbn_k<<<g1d(n_tgt * e4), TB, 0, stream>>>(t1, B.dw, B.dwb, B.g2, B.b2, n_tgt * e4, e4);
            launch_gemm(0, t1, nullptr, 0, B.Wpw, B.bpw, B.g3, B.b3, nullptr,
                        t2, n_tgt, e4, e4, EP_BN_LRELU, stream);
            bn_k<<<g1d(n_tgt * e4), TB, 0, stream>>>(t2, B.g4, B.b4, n_tgt * e4, e4);
            launch_gemm(0, t2, nullptr, 0, B.Wp, B.bp, B.g5, B.b5, xcur,
                        xalt, n_tgt, Cout, e4, EP_BN_ADD_LRELU, stream);
            float* tmp = xcur; xcur = xalt; xalt = tmp;
        }
        se_mean_k<<<g1d(Cout), TB, 0, stream>>>(xcur, zbuf, n_tgt, Cout);
        se_mlp_k<<<1, 256, 0, stream>>>(zbuf, P.seW1, P.seW2, sbuf, Cout);
        scale_k<<<g1d(n_tgt * Cout), TB, 0, stream>>>(xcur, sbuf, n_tgt * Cout, Cout);
        return xcur;
    };

    // ---- SA1..SA4, with downsampled position chain ----
    float* x1 = run_sa(sa[0], x0, p4_0, idx1, e1s, e1t, N1, E1, cins[0], hs[0], couts[0], xA[0], xB[0]);
    gather_rows_k<<<g1d(N1 * 3), TB, 0, stream>>>(pos, idx1, pos1, N1, 3);
    gather_rows_k<<<g1d(N1), TB, 0, stream>>>(refl, idx1, refl1, N1, 1);
    build_pos4_k<<<g1d(N1), TB, 0, stream>>>(pos1, refl1, sf, p4_1, N1);

    float* x2 = run_sa(sa[1], x1, p4_1, idx2, e2s, e2t, N2, E2, cins[1], hs[1], couts[1], xA[1], xB[1]);
    gather_rows_k<<<g1d(N2 * 3), TB, 0, stream>>>(pos1, idx2, pos2, N2, 3);
    gather_rows_k<<<g1d(N2), TB, 0, stream>>>(refl1, idx2, refl2, N2, 1);
    build_pos4_k<<<g1d(N2), TB, 0, stream>>>(pos2, refl2, sf, p4_2, N2);

    float* x3 = run_sa(sa[2], x2, p4_2, idx3, e3s, e3t, N3, E3, cins[2], hs[2], couts[2], xA[2], xB[2]);
    gather_rows_k<<<g1d(N3 * 3), TB, 0, stream>>>(pos2, idx3, pos3, N3, 3);
    gather_rows_k<<<g1d(N3), TB, 0, stream>>>(refl2, idx3, refl3, N3, 1);
    build_pos4_k<<<g1d(N3), TB, 0, stream>>>(pos3, refl3, sf, p4_3, N3);

    float* x4 = run_sa(sa[3], x3, p4_3, idx4, e4s, e4t, N4, E4, cins[3], hs[3], couts[3], xA[3], xB[3]);

    // ---- FP decoder (ping-pong SCRA/SCRB) ----
    concat_k<<<g1d(N3 * 1536), TB, 0, stream>>>(x4, fp4_idx, 1024, x3, 512, SCRA, N3);
    launch_gemm(0, SCRA, nullptr, 0, fp[3].W0, fp[3].b0, nullptr, nullptr, nullptr,
                SCRB, N3, 1024, 1536, EP_LRELU, stream);
    launch_gemm(0, SCRB, nullptr, 0, fp[3].W1, fp[3].b1, fp[3].g, fp[3].beta, nullptr,
                SCRA, N3, 1024, 1024, EP_LRELU_BN, stream);
    float* xf = SCRA;

    struct Stage { const int* idx; const float* skip; int cs; int n; const MlpP* m; };
    Stage stg[3] = { {fp3_idx, x2, 256, N2, &fp[2]},
                     {fp2_idx, x1, 128, N1, &fp[1]},
                     {fp1_idx, x0, 64,  N0, &fp[0]} };
    for (int s = 0; s < 3; ++s) {
        float* other = (xf == SCRA) ? SCRB : SCRA;
        int Cin = 1024 + stg[s].cs;
        concat_k<<<g1d(stg[s].n * Cin), TB, 0, stream>>>(xf, stg[s].idx, 1024,
                                                         stg[s].skip, stg[s].cs, other, stg[s].n);
        launch_gemm(0, other, nullptr, 0, stg[s].m->W0, stg[s].m->b0, nullptr, nullptr, nullptr,
                    xf, stg[s].n, 1024, Cin, EP_LRELU, stream);
        launch_gemm(0, xf, nullptr, 0, stg[s].m->W1, stg[s].m->b1, stg[s].m->g, stg[s].m->beta,
                    nullptr, other, stg[s].n, 1024, 1024, EP_LRELU_BN, stream);
        xf = other;
    }

    // ---- head ----
    float* hh = (xf == SCRA) ? SCRB : SCRA;
    launch_gemm(0, xf, nullptr, 0, hd.W1, hd.b1, hd.g, hd.bb, nullptr,
                hh, N0, 1024, 1024, EP_BN_LRELU, stream);
    logits_k<<<g1d(N0 * 2), TB, 0, stream>>>(hh, hd.W2, hd.b2, (float*)d_out, N0);
}
